// FreqAwareExpertChoiceMoE_46815143526452
// MI455X (gfx1250) — matrix operand order repaired and checked by reading, not compiled, not run on hardware
//
#include <hip/hip_runtime.h>

// ---------------- constants (from reference) ----------------
#define DD    512
#define HH    1024
#define OUTD  512
#define EE    8
#define NBAND 4
#define RRANK 16
#define DCTF  64
#define SCALE_LORA 2.0f      // ALPHA / R = 32 / 16
#define LOSS_COEF_C 0.01f

typedef __attribute__((ext_vector_type(16))) __bf16 v16bf;
typedef __attribute__((ext_vector_type(8)))  float  v8f;

union FragU { unsigned int u[8]; v16bf v; };

__device__ __forceinline__ unsigned short f2bf(float f) {
  unsigned int u = __float_as_uint(f);
  unsigned int r = u + 0x7FFFu + ((u >> 16) & 1u);   // round-to-nearest-even
  return (unsigned short)(r >> 16);
}
__device__ __forceinline__ unsigned int fkey(float f) {
  unsigned int u = __float_as_uint(f);
  return (u & 0x80000000u) ? ~u : (u | 0x80000000u); // monotone float->uint
}
__device__ __forceinline__ float gelu_tanh(float v) {
  float v3 = v * v * v;
  return 0.5f * v * (1.f + tanhf(0.7978845608f * (v + 0.044715f * v3)));
}

// A-fragment (16x32 bf16) from LDS, row-major with stride ldk (elements).
// Layout (ISA 7.12.2): lane L row = L&15; VGPR v holds K = 2*(v&3) + 8*(L>>4) + 16*(v>>2), +{0,1}
__device__ __forceinline__ v16bf load_a_lds(const unsigned short* base, int ldk, int lane) {
  int row = lane & 15, half = lane >> 4;
  const unsigned short* rp = base + row * ldk;
  FragU f;
#pragma unroll
  for (int v = 0; v < 8; ++v) {
    int kk = ((v & 3) << 1) + (half << 3) + ((v >> 2) << 4);
    f.u[v] = *(const unsigned int*)(rp + kk);
  }
  return f.v;
}

// B-fragment from pre-packed global weights:
// packed[( (kc*nch + nc)*32 + lane )*8 + v] = bf16pair( W[k=kc*32+2*(v&3)+8*(lane>>4)+16*(v>>2)][col=nc*16+(lane&15)], W[k+1][col] )
__device__ __forceinline__ v16bf load_b_pack(const unsigned int* base, int kc, int nc, int nch, int lane) {
  const uint4* q = (const uint4*)(base + (((size_t)(kc * nch + nc)) * 32 + lane) * 8);
  uint4 q0 = q[0], q1 = q[1];
  FragU f;
  f.u[0] = q0.x; f.u[1] = q0.y; f.u[2] = q0.z; f.u[3] = q0.w;
  f.u[4] = q1.x; f.u[5] = q1.y; f.u[6] = q1.z; f.u[7] = q1.w;
  return f.v;
}

// ---------------- weight repack to bf16 B-fragment layout ----------------
// mode 0:W1[E,513,1024](rows 0..511)  1:W2[E,1024,512]  2:A1[E,4,513,16]->[512,64]
// mode 3:A2[E,4,1024,16]->[1024,64]   4:B1[E,64,1024]   5:B2[E,64,512]
__global__ __launch_bounds__(256) void pack_kernel(int mode, const float* __restrict__ src,
                                                   unsigned int* __restrict__ dst, int kch, int nch) {
  long long gid = (long long)blockIdx.x * blockDim.x + threadIdx.x;
  long long total = (long long)EE * kch * nch * 256;
  if (gid >= total) return;
  int v = (int)(gid & 7);
  int lane = (int)((gid >> 3) & 31);
  long long rest = gid >> 8;
  int nc = (int)(rest % nch); rest /= nch;
  int kc = (int)(rest % kch);
  int e  = (int)(rest / kch);
  int col = nc * 16 + (lane & 15);
  int kf  = kc * 32 + 2 * (v & 3) + ((lane >> 4) << 3) + ((v >> 2) << 4);
  long long i0 = 0, stride = 0;
  switch (mode) {
    case 0: i0 = ((long long)(e * 513 + kf)) * 1024 + col; stride = 1024; break;
    case 1: i0 = ((long long)(e * 1024 + kf)) * 512 + col; stride = 512;  break;
    case 2: { int b = col >> 4, r = col & 15;
              i0 = (((long long)(e * NBAND + b)) * 513 + kf) * 16 + r; stride = 16; } break;
    case 3: { int b = col >> 4, r = col & 15;
              i0 = (((long long)(e * NBAND + b)) * 1024 + kf) * 16 + r; stride = 16; } break;
    case 4: i0 = ((long long)(e * 64 + kf)) * 1024 + col; stride = 1024; break;
    default:i0 = ((long long)(e * 64 + kf)) * 512 + col; stride = 512;  break;
  }
  unsigned int lo = f2bf(src[i0]);
  unsigned int hi = f2bf(src[i0 + stride]);
  dst[gid] = lo | (hi << 16);
}

// ---------------- gate: W_eff = W_gate[:512] + W_dct @ W_gate[512:] ----------------
__global__ __launch_bounds__(256) void gate_combine_kernel(const float* __restrict__ Wdct,
                                                           const float* __restrict__ Wgate,
                                                           float* __restrict__ Weff) {
  int gid = blockIdx.x * 256 + threadIdx.x;
  if (gid >= DD * EE) return;
  int d = gid >> 3, e = gid & 7;
  float acc = Wgate[d * EE + e];
  for (int c = 0; c < DCTF; ++c)
    acc += Wdct[d * DCTF + c] * Wgate[(DD + c) * EE + e];
  Weff[d * EE + e] = acc;
}

// ---------------- affinity: aff = x @ W_eff ----------------
__global__ __launch_bounds__(256) void affinity_kernel(const float* __restrict__ x,
                                                       const float* __restrict__ Weff,
                                                       float* __restrict__ aff,
                                                       float* __restrict__ affT, int N) {
  __shared__ float wl[DD * EE];
  for (int i = threadIdx.x; i < DD * EE; i += 256) wl[i] = Weff[i];
  __syncthreads();
  int t = blockIdx.x * 256 + threadIdx.x;
  if (t >= N) return;
  float acc[EE];
#pragma unroll
  for (int e = 0; e < EE; ++e) acc[e] = 0.f;
  const float4* xr = (const float4*)(x + (size_t)t * DD);
  for (int i = 0; i < DD / 4; ++i) {
    float4 v = xr[i];
    float xv4[4] = {v.x, v.y, v.z, v.w};
#pragma unroll
    for (int c = 0; c < 4; ++c) {
      int d = i * 4 + c; float xv = xv4[c];
#pragma unroll
      for (int e = 0; e < EE; ++e) acc[e] += xv * wl[d * EE + e];
    }
  }
#pragma unroll
  for (int e = 0; e < EE; ++e) {
    aff[(size_t)t * EE + e] = acc[e];
    affT[(size_t)e * N + t] = acc[e];
  }
}

// ---------------- small init ----------------
__global__ void init_small_kernel(unsigned int* cntGt, unsigned int* cntEq,
                                  unsigned int* loadExtra, float* importance) {
  int i = threadIdx.x;
  if (i < EE) { cntGt[i] = 0u; cntEq[i] = 0u; loadExtra[i] = 0u; importance[i] = 0.f; }
}

// ---------------- per-expert k-th largest threshold (bit binary search) ----------------
__global__ __launch_bounds__(1024) void topk_threshold_kernel(const float* __restrict__ affT,
                                                              int N, int k, unsigned int* __restrict__ T) {
  __shared__ unsigned int red[1024];
  int e = blockIdx.x, tid = threadIdx.x;
  const float* col = affT + (size_t)e * N;
  unsigned long long lo = 0ull, hi = 0xFFFFFFFFull;
  while (lo < hi) {
    unsigned long long mid = (lo + hi + 1ull) >> 1;
    unsigned int m = (unsigned int)mid, cnt = 0;
    for (int t = tid; t < N; t += 1024) cnt += (fkey(col[t]) >= m) ? 1u : 0u;
    red[tid] = cnt; __syncthreads();
    for (int s = 512; s > 0; s >>= 1) { if (tid < s) red[tid] += red[tid + s]; __syncthreads(); }
    unsigned int tot = red[0]; __syncthreads();
    if (tot >= (unsigned int)k) lo = mid; else hi = mid - 1ull;
  }
  if (tid == 0) T[e] = (unsigned int)lo;
}

__global__ __launch_bounds__(256) void compact_gt_kernel(const float* __restrict__ affT,
                                                         const unsigned int* __restrict__ T,
                                                         unsigned int* __restrict__ cntGt,
                                                         int* __restrict__ idxList,
                                                         unsigned int* __restrict__ maskw, int N, int k) {
  int t = blockIdx.x * 256 + threadIdx.x;
  if (t >= N) return;
  unsigned int m = 0;
#pragma unroll
  for (int e = 0; e < EE; ++e) {
    unsigned int kk = fkey(affT[(size_t)e * N + t]);
    if (kk > T[e]) {
      unsigned int p = atomicAdd(&cntGt[e], 1u);
      idxList[(size_t)e * k + p] = t;
      m |= (1u << e);
    }
  }
  maskw[t] = m;
}

__global__ __launch_bounds__(256) void compact_eq_kernel(const float* __restrict__ affT,
                                                         const unsigned int* __restrict__ T,
                                                         const unsigned int* __restrict__ cntGt,
                                                         unsigned int* __restrict__ cntEq,
                                                         int* __restrict__ idxList,
                                                         unsigned int* __restrict__ maskw, int N, int k) {
  int t = blockIdx.x * 256 + threadIdx.x;
  if (t >= N) return;
  unsigned int m = maskw[t];
#pragma unroll
  for (int e = 0; e < EE; ++e) {
    unsigned int kk = fkey(affT[(size_t)e * N + t]);
    if (kk == T[e]) {
      unsigned int p = atomicAdd(&cntEq[e], 1u);
      unsigned int pos = cntGt[e] + p;
      if (pos < (unsigned int)k) { idxList[(size_t)e * k + pos] = t; m |= (1u << e); }
    }
  }
  maskw[t] = m;
}

// ---------------- gating softmax + importance/load stats ----------------
__global__ __launch_bounds__(256) void gating_kernel(const float* __restrict__ aff,
                                                     const unsigned int* __restrict__ maskw,
                                                     float* __restrict__ gw,
                                                     float* __restrict__ importance,
                                                     unsigned int* __restrict__ loadExtra, int N) {
  __shared__ float sImp[EE];
  __shared__ unsigned int sLoad[EE];
  if (threadIdx.x < EE) { sImp[threadIdx.x] = 0.f; sLoad[threadIdx.x] = 0u; }
  __syncthreads();
  int t = blockIdx.x * 256 + threadIdx.x;
  if (t < N) {
    float a[EE];
#pragma unroll
    for (int e = 0; e < EE; ++e) a[e] = aff[(size_t)t * EE + e];
    unsigned int m = maskw[t];
    if (m == 0u) {               // uncovered -> best expert (affects stats only)
      int best = 0; float bv = a[0];
#pragma unroll
      for (int e = 1; e < EE; ++e) if (a[e] > bv) { bv = a[e]; best = e; }
      atomicAdd(&sLoad[best], 1u);
      m = (1u << best);
    }
    float mx = -3.4e38f;
#pragma unroll
    for (int e = 0; e < EE; ++e) if (m & (1u << e)) mx = fmaxf(mx, a[e]);
    float den = 0.f, g[EE];
#pragma unroll
    for (int e = 0; e < EE; ++e) {
      g[e] = (m & (1u << e)) ? expf(a[e] - mx) : 0.f;
      den += g[e];
    }
    float inv = 1.f / den;
#pragma unroll
    for (int e = 0; e < EE; ++e) {
      float v = g[e] * inv;
      gw[(size_t)t * EE + e] = v;
      if (m & (1u << e)) atomicAdd(&sImp[e], v);
    }
  }
  __syncthreads();
  if (threadIdx.x < EE) {
    atomicAdd(&importance[threadIdx.x], sImp[threadIdx.x]);
    atomicAdd(&loadExtra[threadIdx.x], sLoad[threadIdx.x]);
  }
}

// ---------------- zero output ----------------
__global__ void zero_kernel(float* __restrict__ p, long long n) {
  long long i = (long long)blockIdx.x * blockDim.x + threadIdx.x;
  long long st = (long long)gridDim.x * blockDim.x;
  for (; i < n; i += st) p[i] = 0.f;
}

// ---------------- fused expert FFN (WMMA bf16) ----------------
#define XS_LD 520     // 512 + 8 pad (bank-conflict free: 260 dwords % 64 = 4)
#define HS_LD 1032    // 1024 + 8
#define UA_LD 65
#define UB_LD 72      // 64 + 8

__global__ __launch_bounds__(256) void expert_ffn_kernel(
    const float* __restrict__ x, const int* __restrict__ band,
    const int* __restrict__ idxList, const float* __restrict__ gw,
    const unsigned int* __restrict__ W1p, const unsigned int* __restrict__ W2p,
    const unsigned int* __restrict__ A1p, const unsigned int* __restrict__ A2p,
    const unsigned int* __restrict__ B1p, const unsigned int* __restrict__ B2p,
    const float* __restrict__ W1f, const float* __restrict__ b1f,
    const float* __restrict__ A1f, const float* __restrict__ b2f,
    float* __restrict__ outF, int k) {
  __shared__ __align__(16) unsigned short xs[32 * XS_LD];   // x tile bf16
  __shared__ __align__(16) unsigned short hs[32 * HS_LD];   // hidden tile bf16
  __shared__ __align__(16) float          uall[32 * UA_LD]; // LoRA u (all bands) f32
  __shared__ __align__(16) unsigned short ubig[32 * UB_LD]; // band-masked scaled u, bf16
  __shared__ int   tIdx[32];
  __shared__ int   tBand[32];
  __shared__ float tBv[32];
  __shared__ float tGate[32];

  const int e = blockIdx.y;
  const int tid = threadIdx.x;
  const int lane = tid & 31;
  const int wv = tid >> 5;
  const int half = lane >> 4;
  const int l15 = lane & 15;

  const unsigned int* w1p = W1p + (size_t)e * (16 * 64 * 256);
  const unsigned int* w2p = W2p + (size_t)e * (32 * 32 * 256);
  const unsigned int* a1p = A1p + (size_t)e * (16 * 4 * 256);
  const unsigned int* a2p = A2p + (size_t)e * (32 * 4 * 256);
  const unsigned int* b1p = B1p + (size_t)e * (2 * 64 * 256);
  const unsigned int* b2p = B2p + (size_t)e * (2 * 32 * 256);

  if (tid < 32) {
    int slot = blockIdx.x * 32 + tid;
    int ok = slot < k;
    int tok = idxList[(size_t)e * k + (ok ? slot : 0)];
    int b = band[tok] & (NBAND - 1);
    tIdx[tid] = tok; tBand[tid] = b; tBv[tid] = (float)b;
    tGate[tid] = ok ? gw[(size_t)tok * EE + e] : 0.f;
  }
  __syncthreads();

  { // gather x rows, convert to bf16
    int row = tid >> 3;
    int off = (tid & 7) * 64;
    const float4* xr = (const float4*)(x + (size_t)tIdx[row] * DD + off);
    unsigned short* dst = xs + row * XS_LD + off;
#pragma unroll
    for (int i = 0; i < 16; ++i) {
      float4 v = xr[i];
      dst[i * 4 + 0] = f2bf(v.x); dst[i * 4 + 1] = f2bf(v.y);
      dst[i * 4 + 2] = f2bf(v.z); dst[i * 4 + 3] = f2bf(v.w);
    }
  }
  __syncthreads();

  { // u1_all = xs @ A1  (32x64, K=512): one 16x16 tile per wave
    int mt = wv & 1, nt = wv >> 1;
    v8f c = {};
    for (int kc = 0; kc < 16; ++kc) {
      v16bf a = load_a_lds(xs + (mt * 16) * XS_LD + kc * 32, XS_LD, lane);
      v16bf b = load_b_pack(a1p, kc, nt, 4, lane);
      c = __builtin_amdgcn_wmma_f32_16x16x32_bf16(false, a, false, b, (short)0, c, false, false);
    }
#pragma unroll
    for (int v = 0; v < 8; ++v)
      uall[(mt * 16 + v + 8 * half) * UA_LD + nt * 16 + l15] = c[v];
  }
  __syncthreads();

  { // ubig1: select own band, add band-row of A1, fold LoRA scale
    int tk = tid >> 3, cpart = tid & 7;
    int b = tBand[tk]; float bv = tBv[tk];
#pragma unroll
    for (int j = 0; j < 8; ++j) {
      int col = cpart * 8 + j; int cb = col >> 4, r = col & 15;
      float val = 0.f;
      if (cb == b) {
        float u = uall[tk * UA_LD + col] +
                  bv * A1f[(((size_t)(e * NBAND + b)) * 513 + 512) * 16 + r];
        val = SCALE_LORA * u;
      }
      ubig[tk * UB_LD + col] = f2bf(val);
    }
  }
  __syncthreads();

  // layer 1: h = gelu(xs@W1 + ubig@B1 + b1 + bv*W1[512])
  for (int it = 0; it < 16; ++it) {
    int id = wv + it * 8;
    int mt = id & 1, nt = id >> 1;       // nt 0..63
    v8f c = {};
    for (int kc = 0; kc < 16; ++kc) {
      v16bf a = load_a_lds(xs + (mt * 16) * XS_LD + kc * 32, XS_LD, lane);
      v16bf b = load_b_pack(w1p, kc, nt, 64, lane);
      c = __builtin_amdgcn_wmma_f32_16x16x32_bf16(false, a, false, b, (short)0, c, false, false);
    }
#pragma unroll
    for (int kc = 0; kc < 2; ++kc) {
      v16bf a = load_a_lds(ubig + (mt * 16) * UB_LD + kc * 32, UB_LD, lane);
      v16bf b = load_b_pack(b1p, kc, nt, 64, lane);
      c = __builtin_amdgcn_wmma_f32_16x16x32_bf16(false, a, false, b, (short)0, c, false, false);
    }
#pragma unroll
    for (int v = 0; v < 8; ++v) {
      int tr = mt * 16 + v + 8 * half;
      int j = nt * 16 + l15;
      float val = c[v] + b1f[e * HH + j] +
                  tBv[tr] * W1f[((size_t)e * 513 + 512) * 1024 + j];
      hs[tr * HS_LD + j] = f2bf(gelu_tanh(val));
    }
  }
  __syncthreads();

  { // u2_all = hs @ A2  (32x64, K=1024)
    int mt = wv & 1, nt = wv >> 1;
    v8f c = {};
    for (int kc = 0; kc < 32; ++kc) {
      v16bf a = load_a_lds(hs + (mt * 16) * HS_LD + kc * 32, HS_LD, lane);
      v16bf b = load_b_pack(a2p, kc, nt, 4, lane);
      c = __builtin_amdgcn_wmma_f32_16x16x32_bf16(false, a, false, b, (short)0, c, false, false);
    }
#pragma unroll
    for (int v = 0; v < 8; ++v)
      uall[(mt * 16 + v + 8 * half) * UA_LD + nt * 16 + l15] = c[v];
  }
  __syncthreads();

  { // ubig2 (no band-row term: layer-2 input has no band column)
    int tk = tid >> 3, cpart = tid & 7;
    int b = tBand[tk];
#pragma unroll
    for (int j = 0; j < 8; ++j) {
      int col = cpart * 8 + j; int cb = col >> 4;
      float val = (cb == b) ? SCALE_LORA * uall[tk * UA_LD + col] : 0.f;
      ubig[tk * UB_LD + col] = f2bf(val);
    }
  }
  __syncthreads();

  // layer 2: o = hs@W2 + ubig2@B2 + b2, weight by gate, scatter-add
  for (int it = 0; it < 8; ++it) {
    int id = wv + it * 8;
    int mt = id & 1, nt = id >> 1;       // nt 0..31
    v8f c = {};
    for (int kc = 0; kc < 32; ++kc) {
      v16bf a = load_a_lds(hs + (mt * 16) * HS_LD + kc * 32, HS_LD, lane);
      v16bf b = load_b_pack(w2p, kc, nt, 32, lane);
      c = __builtin_amdgcn_wmma_f32_16x16x32_bf16(false, a, false, b, (short)0, c, false, false);
    }
#pragma unroll
    for (int kc = 0; kc < 2; ++kc) {
      v16bf a = load_a_lds(ubig + (mt * 16) * UB_LD + kc * 32, UB_LD, lane);
      v16bf b = load_b_pack(b2p, kc, nt, 32, lane);
      c = __builtin_amdgcn_wmma_f32_16x16x32_bf16(false, a, false, b, (short)0, c, false, false);
    }
#pragma unroll
    for (int v = 0; v < 8; ++v) {
      int tr = mt * 16 + v + 8 * half;
      int j = nt * 16 + l15;
      float val = (c[v] + b2f[e * OUTD + j]) * tGate[tr];
      atomicAdd(&outF[(size_t)tIdx[tr] * OUTD + j], val);
    }
  }
}

// ---------------- loss ----------------
__device__ float cv_sq(const float* v) {
  float m = 0.f;
  for (int e = 0; e < EE; ++e) m += v[e];
  m *= (1.f / EE);
  float s = 0.f;
  for (int e = 0; e < EE; ++e) { float d = v[e] - m; s += d * d; }
  s *= (1.f / (EE - 1));          // ddof=1
  return s / (m * m + 1e-10f);
}
__global__ void loss_kernel(const float* __restrict__ importance,
                            const unsigned int* __restrict__ loadExtra,
                            int k, float* __restrict__ outLoss) {
  if (threadIdx.x != 0 || blockIdx.x != 0) return;
  float a[EE], b[EE];
  for (int e = 0; e < EE; ++e) { a[e] = importance[e]; b[e] = (float)k + (float)loadExtra[e]; }
  *outLoss = (cv_sq(a) + cv_sq(b)) * LOSS_COEF_C;
}

// ---------------- host launcher ----------------
extern "C" void kernel_launch(void* const* d_in, const int* in_sizes, int n_in,
                              void* d_out, int out_size, void* d_ws, size_t ws_size,
                              hipStream_t stream) {
  const float* x     = (const float*)d_in[0];
  const int*   bandI = (const int*)d_in[1];
  const float* Wdct  = (const float*)d_in[2];
  const float* Wgate = (const float*)d_in[3];
  const float* W1    = (const float*)d_in[4];
  const float* b1    = (const float*)d_in[5];
  const float* A1    = (const float*)d_in[6];
  const float* B1    = (const float*)d_in[7];
  const float* W2    = (const float*)d_in[8];
  const float* b2    = (const float*)d_in[9];
  const float* A2    = (const float*)d_in[10];
  const float* B2    = (const float*)d_in[11];
  float* outF = (float*)d_out;

  const int N = in_sizes[0] / DD;
  int cap = (int)((double)N / (double)EE * 1.25);
  if (cap < 1) cap = 1;
  const int k = cap < N ? cap : N;

  // workspace carve
  char* p = (char*)d_ws;
  auto carve = [&](size_t bytes) -> void* {
    void* r = (void*)p;
    p += (bytes + 255) & ~(size_t)255;
    return r;
  };
  float*        aff   = (float*)carve((size_t)N * EE * 4);
  float*        affT  = (float*)carve((size_t)N * EE * 4);
  float*        gwA   = (float*)carve((size_t)N * EE * 4);
  unsigned int* maskw = (unsigned int*)carve((size_t)N * 4);
  float*        Weff  = (float*)carve((size_t)DD * EE * 4);
  unsigned int* Tthr  = (unsigned int*)carve(EE * 4);
  unsigned int* cntGt = (unsigned int*)carve(EE * 4);
  unsigned int* cntEq = (unsigned int*)carve(EE * 4);
  unsigned int* loadX = (unsigned int*)carve(EE * 4);
  float*        impA  = (float*)carve(EE * 4);
  int*          idxL  = (int*)carve((size_t)EE * k * 4);
  unsigned int* W1p   = (unsigned int*)carve((size_t)EE * 16 * 64 * 256 * 4);
  unsigned int* W2p   = (unsigned int*)carve((size_t)EE * 32 * 32 * 256 * 4);
  unsigned int* A1p   = (unsigned int*)carve((size_t)EE * 16 * 4 * 256 * 4);
  unsigned int* A2p   = (unsigned int*)carve((size_t)EE * 32 * 4 * 256 * 4);
  unsigned int* B1p   = (unsigned int*)carve((size_t)EE * 2 * 64 * 256 * 4);
  unsigned int* B2p   = (unsigned int*)carve((size_t)EE * 2 * 32 * 256 * 4);

  // 1) repack weights to bf16 fragment layout
  pack_kernel<<<(EE * 16 * 64 * 256) / 256, 256, 0, stream>>>(0, W1, W1p, 16, 64);
  pack_kernel<<<(EE * 32 * 32 * 256) / 256, 256, 0, stream>>>(1, W2, W2p, 32, 32);
  pack_kernel<<<(EE * 16 * 4  * 256) / 256, 256, 0, stream>>>(2, A1, A1p, 16, 4);
  pack_kernel<<<(EE * 32 * 4  * 256) / 256, 256, 0, stream>>>(3, A2, A2p, 32, 4);
  pack_kernel<<<(EE * 2  * 64 * 256) / 256, 256, 0, stream>>>(4, B1, B1p, 2, 64);
  pack_kernel<<<(EE * 2  * 32 * 256) / 256, 256, 0, stream>>>(5, B2, B2p, 2, 32);

  // 2) gating path
  gate_combine_kernel<<<(DD * EE + 255) / 256, 256, 0, stream>>>(Wdct, Wgate, Weff);
  affinity_kernel<<<(N + 255) / 256, 256, 0, stream>>>(x, Weff, aff, affT, N);
  init_small_kernel<<<1, 32, 0, stream>>>(cntGt, cntEq, loadX, impA);
  topk_threshold_kernel<<<EE, 1024, 0, stream>>>(affT, N, k, Tthr);
  compact_gt_kernel<<<(N + 255) / 256, 256, 0, stream>>>(affT, Tthr, cntGt, idxL, maskw, N, k);
  compact_eq_kernel<<<(N + 255) / 256, 256, 0, stream>>>(affT, Tthr, cntGt, cntEq, idxL, maskw, N, k);
  gating_kernel<<<(N + 255) / 256, 256, 0, stream>>>(aff, maskw, gwA, impA, loadX, N);

  // 3) output
  zero_kernel<<<8192, 256, 0, stream>>>(outF, (long long)N * OUTD + 1);
  dim3 grid((k + 31) / 32, EE);
  expert_ffn_kernel<<<grid, 256, 0, stream>>>(x, bandI, idxL, gwA,
                                              W1p, W2p, A1p, A2p, B1p, B2p,
                                              W1, b1, A1, b2, outF, k);
  loss_kernel<<<1, 1, 0, stream>>>(impA, loadX, k, outF + (size_t)N * OUTD);
}
